// MoEFairScaleFFN_2774548873702
// MI455X (gfx1250) — matrix operand order repaired and checked by reading, not compiled, not run on hardware
//
#include <hip/hip_runtime.h>
#include <math.h>

#define NUM_EXPERTS 8
#define TOPK 2
#define DIM 2048
#define HID 7168
#define NTOK 2048                    // B*S tokens
#define TRASH_ROW (NTOK * TOPK)      // 4096: scratch h-row for padding lanes

typedef __bf16 bf16x16 __attribute__((ext_vector_type(16)));
typedef float  f32x8   __attribute__((ext_vector_type(8)));

union BFrag { uint4 q[2]; bf16x16 v; };

__device__ __forceinline__ unsigned short f32_to_bf16(float f) {
  unsigned u = __float_as_uint(f);
  u += 0x7FFFu + ((u >> 16) & 1u);           // round-to-nearest-even
  return (unsigned short)(u >> 16);
}

#if __has_builtin(__builtin_amdgcn_cvt_pk_bf16_f32)
typedef __bf16 bf16x2 __attribute__((ext_vector_type(2)));
__device__ __forceinline__ unsigned pack_bf16x2(float lo, float hi) {
  union { bf16x2 v; unsigned u; } c;
  c.v = __builtin_amdgcn_cvt_pk_bf16_f32(lo, hi);   // v_cvt_pk_bf16_f32 (RNE)
  return c.u;
}
#else
__device__ __forceinline__ unsigned pack_bf16x2(float lo, float hi) {
  return (unsigned)f32_to_bf16(lo) | ((unsigned)f32_to_bf16(hi) << 16);
}
#endif

// ---------------------------------------------------------------------------
// Kernel 0: init routing lists with harmless padding + zero counters
// ---------------------------------------------------------------------------
__global__ void moe_init(int* __restrict__ cnt, int* __restrict__ tokL,
                         float* __restrict__ wL, int* __restrict__ aidL) {
  int i = blockIdx.x * blockDim.x + threadIdx.x;
  if (i < NUM_EXPERTS) cnt[i] = 0;
  if (i < NUM_EXPERTS * NTOK) {
    tokL[i] = 0;            // padding rows point at token 0 ...
    wL[i]   = 0.0f;         // ... with weight 0 (adds exactly 0.0f)
    aidL[i] = TRASH_ROW;    // ... and write h into a trash row
  }
}

// ---------------------------------------------------------------------------
// Kernel 1: gating — one wave32 per token. scores = x @ Wg, top-2, softmax,
// append (token, weight, assignment-id) to the chosen experts' lists.
// ---------------------------------------------------------------------------
__global__ void moe_gate(const float* __restrict__ x, const float* __restrict__ Wg,
                         int* __restrict__ cnt, int* __restrict__ tokL,
                         float* __restrict__ wL, int* __restrict__ aidL) {
  const int wave = threadIdx.x >> 5;
  const int lane = threadIdx.x & 31;
  const int t = blockIdx.x * 8 + wave;
  if (t >= NTOK) return;

  float s[NUM_EXPERTS];
#pragma unroll
  for (int e = 0; e < NUM_EXPERTS; ++e) s[e] = 0.0f;

  const float* xr = x + (size_t)t * DIM;
  for (int d = lane; d < DIM; d += 32) {
    const float xv = xr[d];
    const float* g = Wg + (size_t)d * NUM_EXPERTS;
#pragma unroll
    for (int e = 0; e < NUM_EXPERTS; ++e) s[e] += xv * g[e];
  }
#pragma unroll
  for (int off = 16; off > 0; off >>= 1) {
#pragma unroll
    for (int e = 0; e < NUM_EXPERTS; ++e) s[e] += __shfl_down(s[e], off, 32);
  }

  if (lane == 0) {
    float s0 = -3.4e38f, s1 = -3.4e38f;
    int i0 = 0, i1 = 0;
#pragma unroll
    for (int e = 0; e < NUM_EXPERTS; ++e) {
      if (s[e] > s0) { s1 = s0; i1 = i0; s0 = s[e]; i0 = e; }
      else if (s[e] > s1) { s1 = s[e]; i1 = e; }
    }
    // softmax over the two selected scores
    const float e1 = __expf(s1 - s0);
    const float inv = __builtin_amdgcn_rcpf(1.0f + e1);
    const float w0 = inv, w1 = e1 * inv;

    int slot0 = atomicAdd(&cnt[i0], 1);
    int idx0  = i0 * NTOK + slot0;
    tokL[idx0] = t; wL[idx0] = w0; aidL[idx0] = t * TOPK + 0;

    int slot1 = atomicAdd(&cnt[i1], 1);
    int idx1  = i1 * NTOK + slot1;
    tokL[idx1] = t; wL[idx1] = w1; aidL[idx1] = t * TOPK + 1;
  }
}

// ---------------------------------------------------------------------------
// Kernel 2: h = silu(x @ w1[e]) * (x @ w3[e]) for routed rows.
// Block = 256 threads = 8 waves -> 64 tokens x 128 hidden cols per block.
// Wave w: wr = w/4 (32-row band), wc = w%4 (32-col band); each wave owns a
// 2x2 grid of 16x16 accumulators per matrix -> 8 WMMAs per K-step.
// ---------------------------------------------------------------------------
__global__ __launch_bounds__(256) void moe_h(
    const float* __restrict__ x, const float* __restrict__ w1,
    const float* __restrict__ w3, const int* __restrict__ cnt,
    const int* __restrict__ tokL, const int* __restrict__ aidL,
    unsigned short* __restrict__ hbuf) {
  const int e  = blockIdx.z;
  const int mb = blockIdx.y;           // 64-slot block within expert list
  const int nb = blockIdx.x * 128;     // HIDDEN column base
  if (mb * 64 >= cnt[e]) return;       // whole tile is padding -> skip

  __shared__ __align__(16) unsigned short aT[64 * 32];    // x tile  [M][K] bf16
  __shared__ __align__(16) unsigned short b1T[128 * 32];  // w1 tile [N][K] bf16 (transposed)
  __shared__ __align__(16) unsigned short b3T[128 * 32];  // w3 tile [N][K] bf16 (transposed)
  __shared__ int aid_s[64];
  __shared__ int tok_s[64];

  const int tid = threadIdx.x;
  if (tid < 64) {
    const int gi = e * NTOK + mb * 64 + tid;
    aid_s[tid] = aidL[gi];
    tok_s[tid] = tokL[gi];
  }
  __syncthreads();

  const float* w1e = w1 + (size_t)e * DIM * HID;
  const float* w3e = w3 + (size_t)e * DIM * HID;

  const int lane = tid & 31;
  const int wv   = tid >> 5;
  const int wr   = wv >> 2;            // 0..1  (32-row band)
  const int wc   = wv & 3;             // 0..3  (32-col band)
  const int half = lane >> 4;
  const int l16  = lane & 15;

  f32x8 acc1[2][2] = {};
  f32x8 acc3[2][2] = {};

  // staging assignments
  const int ar = tid >> 2;             // A: row 0..63
  const int ac = (tid & 3) * 8;        // A: 8 consecutive K
  const int bn = tid & 127;            // B: column n
  const int kp = (tid >> 7) * 2;       // B: k-pair phase {0,2}

  // hoist loop-invariant staging bases out of the K loop
  const float* xrowA = x + (size_t)tok_s[ar] * DIM + ac;          // one LDS read, once
  const float* b1base = w1e + (size_t)kp * HID + nb + bn;
  const float* b3base = w3e + (size_t)kp * HID + nb + bn;

  unsigned* b1w = (unsigned*)b1T;
  unsigned* b3w = (unsigned*)b3T;

  for (int k0 = 0; k0 < DIM; k0 += 32) {
    { // stage A: fp32 -> bf16, one 16B LDS store per thread
      const float4* xr4 = (const float4*)(xrowA + k0);
      const float4 v0 = xr4[0];
      const float4 v1 = xr4[1];
      uint4 pk;
      pk.x = pack_bf16x2(v0.x, v0.y);
      pk.y = pack_bf16x2(v0.z, v0.w);
      pk.z = pack_bf16x2(v1.x, v1.y);
      pk.w = pack_bf16x2(v1.z, v1.w);
      ((uint4*)aT)[ar * 4 + (ac >> 3)] = pk;
    }
#pragma unroll
    for (int kk = 0; kk < 8; ++kk) {   // stage B1/B3 transposed [N][K], dword stores
      const int k = kp + kk * 4;       // covers even k 0..30 across the two phases
      const size_t g0 = (size_t)(k0 + kk * 4) * HID;
      b1w[bn * 16 + (k >> 1)] = pack_bf16x2(b1base[g0], b1base[g0 + HID]);
      b3w[bn * 16 + (k >> 1)] = pack_bf16x2(b3base[g0], b3base[g0 + HID]);
      if (kk == 0 && k0 + 32 < DIM) {  // prefetch next K-step's weight rows
        __builtin_prefetch(&b1base[g0 + (size_t)32 * HID], 0, 0);
        __builtin_prefetch(&b3base[g0 + (size_t)32 * HID], 0, 0);
      }
    }
    __syncthreads();

    BFrag aF[2], b1F[2], b3F[2];
#pragma unroll
    for (int i = 0; i < 2; ++i) {
      // A frag: lane half h holds K = h*16..h*16+15 of its row
      const uint4* ap = (const uint4*)&aT[(wr * 32 + i * 16 + l16) * 32];
      aF[i].q[0] = ap[half * 2];
      aF[i].q[1] = ap[half * 2 + 1];
      // B frag: lane half h holds K = h*16..h*16+15 of its column
      const uint4* q1 = (const uint4*)&b1T[(wc * 32 + i * 16 + l16) * 32];
      b1F[i].q[0] = q1[half * 2];
      b1F[i].q[1] = q1[half * 2 + 1];
      const uint4* q3 = (const uint4*)&b3T[(wc * 32 + i * 16 + l16) * 32];
      b3F[i].q[0] = q3[half * 2];
      b3F[i].q[1] = q3[half * 2 + 1];
    }
#pragma unroll
    for (int mi = 0; mi < 2; ++mi) {
#pragma unroll
      for (int ni = 0; ni < 2; ++ni) {
        acc1[mi][ni] = __builtin_amdgcn_wmma_f32_16x16x32_bf16(
            false, aF[mi].v, false, b1F[ni].v, (short)0, acc1[mi][ni], false, false);
        acc3[mi][ni] = __builtin_amdgcn_wmma_f32_16x16x32_bf16(
            false, aF[mi].v, false, b3F[ni].v, (short)0, acc3[mi][ni], false, false);
      }
    }
    __syncthreads();
  }

  // epilogue: SwiGLU (silu via fast rcp), store bf16 h row-by-assignment
#pragma unroll
  for (int mi = 0; mi < 2; ++mi) {
#pragma unroll
    for (int ni = 0; ni < 2; ++ni) {
      const int col = nb + wc * 32 + ni * 16 + l16;
#pragma unroll
      for (int j = 0; j < 8; ++j) {
        const int m = wr * 32 + mi * 16 + j + half * 8;  // C/D: VGPR j -> M=j (+8 upper half)
        const float z = acc1[mi][ni][j];
        const float sig = __builtin_amdgcn_rcpf(1.0f + __expf(-z));
        const float h = z * sig * acc3[mi][ni][j];
        hbuf[(size_t)aid_s[m] * HID + col] = f32_to_bf16(h);
      }
    }
  }
}

// ---------------------------------------------------------------------------
// Kernel 3: y[t] += weight * (h @ w2[e]) for routed rows (atomic combine).
// Same 64x128 tiling; K = HIDDEN = 7168; A already bf16 in hbuf.
// ---------------------------------------------------------------------------
__global__ __launch_bounds__(256) void moe_out(
    const unsigned short* __restrict__ hbuf, const float* __restrict__ w2,
    const int* __restrict__ cnt, const int* __restrict__ tokL,
    const float* __restrict__ wL, const int* __restrict__ aidL,
    float* __restrict__ y) {
  const int e  = blockIdx.z;
  const int mb = blockIdx.y;
  const int nb = blockIdx.x * 128;     // DIM column base
  if (mb * 64 >= cnt[e]) return;

  __shared__ __align__(16) unsigned short aT[64 * 32];    // h tile  [M][K]
  __shared__ __align__(16) unsigned short bT[128 * 32];   // w2 tile [N][K] (transposed)
  __shared__ int   tok_s[64];
  __shared__ float w_s[64];
  __shared__ int   aid_s[64];

  const int tid = threadIdx.x;
  if (tid < 64) {
    const int gi = e * NTOK + mb * 64 + tid;
    tok_s[tid] = tokL[gi];
    w_s[tid]   = wL[gi];
    aid_s[tid] = aidL[gi];
  }
  __syncthreads();

  const float* w2e = w2 + (size_t)e * HID * DIM;

  const int lane = tid & 31;
  const int wv   = tid >> 5;
  const int wr   = wv >> 2;
  const int wc   = wv & 3;
  const int half = lane >> 4;
  const int l16  = lane & 15;

  f32x8 acc[2][2] = {};

  const int ar = tid >> 2;
  const int ac = (tid & 3) * 8;
  const int bn = tid & 127;
  const int kp = (tid >> 7) * 2;

  // hoisted staging bases
  const unsigned short* arow = hbuf + (size_t)aid_s[ar] * HID + ac;  // one LDS read, once
  const float* bbase = w2e + (size_t)kp * DIM + nb + bn;

  unsigned* bw = (unsigned*)bT;

  for (int k0 = 0; k0 < HID; k0 += 32) {
    { // stage A: h already bf16, straight 16B copy
      const uint4 v = *(const uint4*)(arow + k0);
      ((uint4*)aT)[ar * 4 + (ac >> 3)] = v;
    }
#pragma unroll
    for (int kk = 0; kk < 8; ++kk) {   // stage w2 transposed [N][K], dword stores
      const int k = kp + kk * 4;
      const size_t g0 = (size_t)(k0 + kk * 4) * DIM;
      bw[bn * 16 + (k >> 1)] = pack_bf16x2(bbase[g0], bbase[g0 + DIM]);
      if (kk == 0 && k0 + 32 < HID) {
        __builtin_prefetch(&bbase[g0 + (size_t)32 * DIM], 0, 0);
      }
    }
    __syncthreads();

    BFrag aF[2], bF[2];
#pragma unroll
    for (int i = 0; i < 2; ++i) {
      const uint4* ap = (const uint4*)&aT[(wr * 32 + i * 16 + l16) * 32];
      aF[i].q[0] = ap[half * 2];
      aF[i].q[1] = ap[half * 2 + 1];
      const uint4* bp = (const uint4*)&bT[(wc * 32 + i * 16 + l16) * 32];
      bF[i].q[0] = bp[half * 2];
      bF[i].q[1] = bp[half * 2 + 1];
    }
#pragma unroll
    for (int mi = 0; mi < 2; ++mi) {
#pragma unroll
      for (int ni = 0; ni < 2; ++ni) {
        acc[mi][ni] = __builtin_amdgcn_wmma_f32_16x16x32_bf16(
            false, aF[mi].v, false, bF[ni].v, (short)0, acc[mi][ni], false, false);
      }
    }
    __syncthreads();
  }

#pragma unroll
  for (int mi = 0; mi < 2; ++mi) {
#pragma unroll
    for (int ni = 0; ni < 2; ++ni) {
      const int col = nb + wc * 32 + ni * 16 + l16;
#pragma unroll
      for (int j = 0; j < 8; ++j) {
        const int m = wr * 32 + mi * 16 + j + half * 8;
        // padding rows have w_s == 0 -> contributes exactly 0.0f
        atomicAdd(&y[(size_t)tok_s[m] * DIM + col], w_s[m] * acc[mi][ni][j]);
      }
    }
  }
}

// ---------------------------------------------------------------------------
extern "C" void kernel_launch(void* const* d_in, const int* in_sizes, int n_in,
                              void* d_out, int out_size, void* d_ws, size_t ws_size,
                              hipStream_t stream) {
  const float* x  = (const float*)d_in[0];
  const float* Wg = (const float*)d_in[1];
  const float* w1 = (const float*)d_in[2];
  const float* w3 = (const float*)d_in[3];
  const float* w2 = (const float*)d_in[4];
  float* y = (float*)d_out;

  // workspace layout
  char* ws = (char*)d_ws;
  int*   cnt  = (int*)(ws);                                  // 8 ints (256 B reserved)
  int*   tokL = (int*)(ws + 256);                            // 8*2048 ints
  float* wL   = (float*)(ws + 256 + 1 * 65536);              // 8*2048 floats
  int*   aidL = (int*)(ws + 256 + 2 * 65536);                // 8*2048 ints
  unsigned short* hbuf = (unsigned short*)(ws + 256 + 3 * 65536);
  // hbuf: (4096 + 1 trash) rows x 7168 bf16 ~= 58.7 MB

  hipMemsetAsync(d_out, 0, (size_t)out_size * sizeof(float), stream);

  moe_init<<<(NUM_EXPERTS * NTOK + 255) / 256, 256, 0, stream>>>(cnt, tokL, wL, aidL);
  moe_gate<<<NTOK / 8, 256, 0, stream>>>(x, Wg, cnt, tokL, wL, aidL);
  moe_h<<<dim3(HID / 128, NTOK / 64, NUM_EXPERTS), 256, 0, stream>>>(
      x, w1, w3, cnt, tokL, aidL, hbuf);
  moe_out<<<dim3(DIM / 128, NTOK / 64, NUM_EXPERTS), 256, 0, stream>>>(
      hbuf, w2, cnt, tokL, wL, aidL, y);
}
